// FuseAttention_38345468019064
// MI455X (gfx1250) — matrix operand order repaired and checked by reading, not compiled, not run on hardware
//
#include <hip/hip_runtime.h>
#include <hip/hip_bf16.h>

// ---------------------------------------------------------------------------
// FuseAttention for MI455X (gfx1250, wave32, WMMA bf16 w/ f32 accumulate)
// B=4, C=256, H=W=64, N=4096, d_qk=32, hidden=16
// Flash-attention stage double-buffers K/V tiles in LDS via
// GLOBAL_LOAD_ASYNC_TO_LDS_B128 (ASYNCcnt) when the builtin is available.
// ---------------------------------------------------------------------------

#define BB 4
#define CC 256
#define NN 4096
#define HID 16

typedef __attribute__((ext_vector_type(16))) __bf16 v16bf;
typedef __attribute__((ext_vector_type(8)))  __bf16 v8bf;
typedef __attribute__((ext_vector_type(8)))  float  v8f;
typedef __attribute__((ext_vector_type(4)))  float  f32x4;
typedef __attribute__((ext_vector_type(4)))  int    v4i;

#if __has_builtin(__builtin_amdgcn_global_load_async_to_lds_b128) && \
    __has_builtin(__builtin_amdgcn_s_wait_asynccnt)
#define HAVE_ASYNC 1
#else
#define HAVE_ASYNC 0
#endif

#if HAVE_ASYNC
#define WAIT_ASYNC(n) __builtin_amdgcn_s_wait_asynccnt(n)
#else
#define WAIT_ASYNC(n) ((void)0)
#endif

// copy 16 bytes global -> LDS (async if available)
__device__ __forceinline__ void cp16(void* ldst, const void* gsrc) {
#if HAVE_ASYNC
  __builtin_amdgcn_global_load_async_to_lds_b128(
      (__attribute__((address_space(1))) v4i*)gsrc,
      (__attribute__((address_space(3))) v4i*)ldst, 0, 0);
#else
  *(f32x4*)ldst = *(const f32x4*)gsrc;
#endif
}

__device__ __forceinline__ v8f wmma_bf16(v16bf a, v16bf b, v8f c) {
  // (neg_a, A, neg_b, B, c_mod, C, reuse_a, reuse_b)
  return __builtin_amdgcn_wmma_f32_16x16x32_bf16(false, a, false, b, (short)0, c, false, false);
}

__device__ __forceinline__ float sigm(float x) { return 1.0f / (1.0f + __expf(-x)); }

// Load a 16x32 bf16 A-tile fragment for this lane (rows in global/LDS memory).
// A element (m,k): m = lane&15 (arow already points at that row),
// k = 16*laneHi + i  (i = 0..15).
__device__ __forceinline__ v16bf load_a16(const __bf16* arow, int kc, int laneHi) {
  v8bf lo = *(const v8bf*)(arow + kc * 32 + 16 * laneHi);
  v8bf hi = *(const v8bf*)(arow + kc * 32 + 16 * laneHi + 8);
  v16bf a;
#pragma unroll
  for (int i = 0; i < 8; i++) { a[i] = lo[i]; a[8 + i] = hi[i]; }
  return a;
}

// Load 16 contiguous bf16 (32B) as two 16B pieces (works for 16B-aligned rows).
__device__ __forceinline__ v16bf ld32(const __bf16* p) {
  v8bf lo = *(const v8bf*)p;
  v8bf hi = *(const v8bf*)(p + 8);
  v16bf r;
#pragma unroll
  for (int i = 0; i < 8; i++) { r[i] = lo[i]; r[8 + i] = hi[i]; }
  return r;
}

// ---------------------------------------------------------------------------
// 1) Spatial attention logits: ys[b][n] = sa0*mean_c + sa1*max_c
// ---------------------------------------------------------------------------
__global__ void k_spatial(const float* __restrict__ x, const float* __restrict__ sa_w,
                          float* __restrict__ ys) {
  int idx = blockIdx.x * blockDim.x + threadIdx.x;
  if (idx >= BB * NN) return;
  int b = idx / NN, n = idx % NN;
  const float* xp = x + (size_t)b * CC * NN + n;
  float s = 0.f, m = -__builtin_inff();
  for (int c = 0; c < CC; c++) { float v = xp[(size_t)c * NN]; s += v; m = fmaxf(m, v); }
  ys[idx] = sa_w[0] * (s * (1.0f / CC)) + sa_w[1] * m;
}

// ---------------------------------------------------------------------------
// 2) Channel pooling: cavg/cmax[b][c] over N
// ---------------------------------------------------------------------------
__global__ void k_chanpool(const float* __restrict__ x, float* __restrict__ cavg,
                           float* __restrict__ cmax) {
  int bc = blockIdx.x;  // b*C + c
  const float* xp = x + (size_t)bc * NN;
  float s = 0.f, m = -__builtin_inff();
  for (int i = threadIdx.x; i < NN; i += blockDim.x) { float v = xp[i]; s += v; m = fmaxf(m, v); }
  __shared__ float ss[256], sm[256];
  ss[threadIdx.x] = s; sm[threadIdx.x] = m; __syncthreads();
  for (int st = 128; st > 0; st >>= 1) {
    if ((int)threadIdx.x < st) {
      ss[threadIdx.x] += ss[threadIdx.x + st];
      sm[threadIdx.x] = fmaxf(sm[threadIdx.x], sm[threadIdx.x + st]);
    }
    __syncthreads();
  }
  if (threadIdx.x == 0) { cavg[bc] = ss[0] * (1.0f / NN); cmax[bc] = sm[0]; }
}

// ---------------------------------------------------------------------------
// 3) Channel MLP: yc[b][c] = sum_h w2[c][h]*(relu(w1·avg)[h] + relu(w1·max)[h])
// ---------------------------------------------------------------------------
__global__ void k_camlp(const float* __restrict__ cavg, const float* __restrict__ cmax,
                        const float* __restrict__ w1, const float* __restrict__ w2,
                        float* __restrict__ yc) {
  int b = blockIdx.x;
  __shared__ float hsum[HID];
  __shared__ float ha[HID], hm[HID];
  int t = threadIdx.x;
  if (t < 2 * HID) {
    int h = t % HID;
    const float* pool = (t < HID) ? (cavg + b * CC) : (cmax + b * CC);
    float s = 0.f;
    for (int i = 0; i < CC; i++) s += w1[h * CC + i] * pool[i];
    float r = fmaxf(s, 0.f);
    if (t < HID) ha[h] = r; else hm[h] = r;
  }
  __syncthreads();
  if (t < HID) hsum[t] = ha[t] + hm[t];
  __syncthreads();
  if (t < CC) {
    float s = 0.f;
#pragma unroll
    for (int h = 0; h < HID; h++) s += w2[t * HID + h] * hsum[h];
    yc[b * CC + t] = s;
  }
}

// ---------------------------------------------------------------------------
// 4) Fuse + transpose into pixel-major bf16 zT[b][n][512]
// ---------------------------------------------------------------------------
__global__ void k_fusez(const float* __restrict__ x, const float* __restrict__ ys,
                        const float* __restrict__ yc, __bf16* __restrict__ zT) {
  int b = blockIdx.x, c0 = blockIdx.y * 32, n0 = blockIdx.z * 32;
  __shared__ float lx[32][33], lf[32][33];
  int tx = threadIdx.x, ty = threadIdx.y;
  const float* xb = x + ((size_t)b * CC + c0) * NN + n0;
  float yv = ys[b * NN + n0 + tx];
#pragma unroll
  for (int i = 0; i < 4; i++) {
    int cc = ty + 8 * i;
    float v = xb[(size_t)cc * NN + tx];
    float cv = yc[b * CC + c0 + cc];
    lx[cc][tx] = v;
    lf[cc][tx] = sigm(yv * v) + sigm(cv * v);
  }
  __syncthreads();
  __bf16* zb = zT + (size_t)b * NN * 512;
#pragma unroll
  for (int i = 0; i < 4; i++) {
    int n = n0 + ty + 8 * i;
    zb[(size_t)n * 512 + c0 + tx]       = (__bf16)lx[tx][ty + 8 * i];
    zb[(size_t)n * 512 + 256 + c0 + tx] = (__bf16)lf[tx][ty + 8 * i];
  }
}

// ---------------------------------------------------------------------------
// 5) Weight conversion fp32 -> bf16
// ---------------------------------------------------------------------------
__global__ void k_cvt(const float* s0, __bf16* d0, int n0,
                      const float* s1, __bf16* d1, int n1,
                      const float* s2, __bf16* d2, int n2,
                      const float* s3, __bf16* d3, int n3) {
  int i = blockIdx.x * blockDim.x + threadIdx.x;
  if (i < n0) d0[i] = (__bf16)s0[i];
  if (i < n1) d1[i] = (__bf16)s1[i];
  if (i < n2) d2[i] = (__bf16)s2[i];
  if (i < n3) d3[i] = (__bf16)s3[i];
}

// ---------------------------------------------------------------------------
// 6) PixelAttention GEMM (WMMA): y_p^T = zT[Nx512] x pa_w1^T, scaled by pa_w2.
// ---------------------------------------------------------------------------
__global__ __launch_bounds__(256) void k_gemm_pa(const __bf16* __restrict__ zT,
                                                 const __bf16* __restrict__ wbf,
                                                 const float* __restrict__ pa_w2,
                                                 float* __restrict__ ypf,
                                                 __bf16* __restrict__ ypT) {
  int wave = threadIdx.x >> 5, lane = threadIdx.x & 31;
  int laneHi = lane >> 4, laneLo = lane & 15;
  int t = blockIdx.x * 8 + wave;
  int b = t / 4096; int rem = t % 4096;
  int rt = rem >> 4, ct = rem & 15;
  int pix0 = rt * 16;
  const __bf16* arow = zT + ((size_t)b * NN + pix0 + laneLo) * 512;
  const __bf16* brow = wbf + (size_t)(ct * 16 + laneLo) * 512;
  v8f acc = {};
#pragma unroll
  for (int kc = 0; kc < 16; kc++) {
    v16bf a = load_a16(arow, kc, laneHi);
    v16bf bv = *(const v16bf*)(brow + kc * 32 + 16 * laneHi);
    acc = wmma_bf16(a, bv, acc);
  }
  int co = ct * 16 + laneLo;
  float scale = pa_w2[co];
  float* yo = ypf + ((size_t)b * CC + co) * NN + pix0 + 8 * laneHi;
  __bf16* yt = ypT + ((size_t)b * NN + pix0 + 8 * laneHi) * 256 + co;
#pragma unroll
  for (int r = 0; r < 8; r++) {
    float v = acc[r] * scale;
    yo[r] = v;
    yt[(size_t)r * 256] = (__bf16)v;
  }
}

// ---------------------------------------------------------------------------
// 7) QKV GEMMs (WMMA): ct 0..1 -> qT, 2..3 -> kT, 4..19 -> v[c][N]
// ---------------------------------------------------------------------------
__global__ __launch_bounds__(256) void k_gemm_qkv(const __bf16* __restrict__ ypT,
                                                  const __bf16* __restrict__ wq,
                                                  const __bf16* __restrict__ wk,
                                                  const __bf16* __restrict__ wv,
                                                  const float* __restrict__ qb,
                                                  const float* __restrict__ kb_,
                                                  const float* __restrict__ vb,
                                                  __bf16* __restrict__ qT,
                                                  __bf16* __restrict__ kT,
                                                  __bf16* __restrict__ vm) {
  int wave = threadIdx.x >> 5, lane = threadIdx.x & 31;
  int laneHi = lane >> 4, laneLo = lane & 15;
  int t = blockIdx.x * 8 + wave;
  int b = t / 5120; int rem = t % 5120;
  int rt = rem / 20, ct = rem % 20;
  int pix0 = rt * 16;
  const __bf16* arow = ypT + ((size_t)b * NN + pix0 + laneLo) * 256;
  const __bf16* W; const float* bias; int col;
  if (ct < 2)       { W = wq; bias = qb;  col = ct * 16 + laneLo; }
  else if (ct < 4)  { W = wk; bias = kb_; col = (ct - 2) * 16 + laneLo; }
  else              { W = wv; bias = vb;  col = (ct - 4) * 16 + laneLo; }
  const __bf16* brow = W + (size_t)col * 256;
  v8f acc = {};
#pragma unroll
  for (int kc = 0; kc < 8; kc++) {
    v16bf a = load_a16(arow, kc, laneHi);
    v16bf bv = *(const v16bf*)(brow + kc * 32 + 16 * laneHi);
    acc = wmma_bf16(a, bv, acc);
  }
  float bvl = bias[col];
  if (ct < 4) {
    __bf16* dst = (ct < 2 ? qT : kT) + ((size_t)b * NN + pix0 + 8 * laneHi) * 32 + col;
#pragma unroll
    for (int r = 0; r < 8; r++) dst[(size_t)r * 32] = (__bf16)(acc[r] + bvl);
  } else {
    __bf16* dst = vm + ((size_t)b * CC + col) * NN + pix0 + 8 * laneHi;
    v8bf o;
#pragma unroll
    for (int r = 0; r < 8; r++) o[r] = (__bf16)(acc[r] + bvl);
    *(v8bf*)dst = o;
  }
}

// ---------------------------------------------------------------------------
// 8) Flash attention + epilogue.
//    Block = 8 waves = 64 queries x 256 v-channels. Per key tile (64 keys):
//    K (4KB) + V (32KB) async-staged into double-buffered LDS once per block
//    (ASYNCcnt + s_wait_asynccnt + barrier), consumed via padded
//    bank-conflict-free ds reads. out = gamma*acc/l + y_p.
// ---------------------------------------------------------------------------
#define KP 40   // padded K-tile row (32 data + 8 pad bf16)
#define VP 72   // padded V-tile row (64 data + 8 pad bf16)
#define PP 72   // padded P row (64 data + 8 pad bf16)
#define STAGE_OPS 9

__global__ __launch_bounds__(256) void k_flash(const __bf16* __restrict__ qT,
                                               const __bf16* __restrict__ kTm,
                                               const __bf16* __restrict__ vm,
                                               const float* __restrict__ ypf,
                                               const float* __restrict__ gamma_p,
                                               float* __restrict__ out) {
  int tid = threadIdx.x;
  int wave = tid >> 5, lane = tid & 31;
  int laneHi = lane >> 4, laneLo = lane & 15;
  int r = wave >> 1, h = wave & 1;
  int b = blockIdx.x / (NN / 64);
  int q0 = (blockIdx.x % (NN / 64)) * 64;

  __shared__ __bf16 ldsK[2][64 * KP];    // 2 x 5KB
  __shared__ __bf16 ldsV[2][256 * VP];   // 2 x 36KB
  __shared__ __bf16 plds[8][16 * PP];    // 18KB per-wave P staging
  __bf16* myp = plds[wave];

  const __bf16* kbase = kTm + (size_t)b * NN * 32;
  const __bf16* vbase = vm + (size_t)b * CC * NN;

  // async stage of key-tile j into buffer `buf` (9 x 16B per thread)
  auto stage = [&](int j, int buf) {
    const __bf16* vsrc = vbase + (size_t)tid * NN + j * 64;   // channel = tid
    __bf16* vdst = &ldsV[buf][tid * VP];
#pragma unroll
    for (int i = 0; i < 8; i++) cp16(vdst + i * 8, vsrc + i * 8);
    int krow = tid >> 2, kch = tid & 3;
    cp16(&ldsK[buf][krow * KP + kch * 8],
         kbase + (size_t)(j * 64 + krow) * 32 + kch * 8);
  };

  // Q A-fragment (held for the whole kernel)
  const __bf16* qrow = qT + ((size_t)b * NN + q0 + r * 16 + laneLo) * 32;
  v16bf qa;
  {
    v8bf lo = *(const v8bf*)(qrow + 16 * laneHi);
    v8bf hi = *(const v8bf*)(qrow + 16 * laneHi + 8);
#pragma unroll
    for (int i = 0; i < 8; i++) { qa[i] = lo[i]; qa[8 + i] = hi[i]; }
  }

  float mrow[8], lrow[8];
  v8f acc[8];
#pragma unroll
  for (int i = 0; i < 8; i++) { mrow[i] = -__builtin_inff(); lrow[i] = 0.f; acc[i] = (v8f)(0.f); }

  stage(0, 0);

  const int J = NN / 64;
  for (int j = 0; j < J; j++) {
    int buf = j & 1;
    if (j + 1 < J) {
      stage(j + 1, buf ^ 1);
      WAIT_ASYNC(STAGE_OPS);   // oldest stage (this j) complete
    } else {
      WAIT_ASYNC(0);
    }
    __syncthreads();           // buf is fully staged for all waves

    const __bf16* Kb = &ldsK[buf][0];
    const __bf16* Vb = &ldsV[buf][0];

    // ---- energy tile S[16 x 64] ----
    v8f s[4];
#pragma unroll
    for (int t4 = 0; t4 < 4; t4++) {
      v16bf kb = ld32(Kb + (t4 * 16 + laneLo) * KP + 16 * laneHi);
      v8f z = {};
      s[t4] = wmma_bf16(qa, kb, z);
    }
    // ---- online softmax update (row m = r0 + 8*laneHi) ----
    float scale_r[8];
#pragma unroll
    for (int r0 = 0; r0 < 8; r0++) {
      float tm = fmaxf(fmaxf(s[0][r0], s[1][r0]), fmaxf(s[2][r0], s[3][r0]));
#pragma unroll
      for (int m5 = 1; m5 < 16; m5 <<= 1) tm = fmaxf(tm, __shfl_xor(tm, m5, 32));
      float mnew = fmaxf(mrow[r0], tm);
      float sc = __expf(mrow[r0] - mnew);
      float rs = 0.f;
#pragma unroll
      for (int t4 = 0; t4 < 4; t4++) { float p = __expf(s[t4][r0] - mnew); s[t4][r0] = p; rs += p; }
#pragma unroll
      for (int m5 = 1; m5 < 16; m5 <<= 1) rs += __shfl_xor(rs, m5, 32);
      lrow[r0] = lrow[r0] * sc + rs;
      mrow[r0] = mnew;
      scale_r[r0] = sc;
    }
#pragma unroll
    for (int ct = 0; ct < 8; ct++)
#pragma unroll
      for (int r0 = 0; r0 < 8; r0++) acc[ct][r0] *= scale_r[r0];

    // ---- P: D-layout -> row-major LDS [m][k] (wave-private, DS in-order) ----
#pragma unroll
    for (int t4 = 0; t4 < 4; t4++)
#pragma unroll
      for (int r0 = 0; r0 < 8; r0++)
        myp[(r0 + 8 * laneHi) * PP + t4 * 16 + laneLo] = (__bf16)s[t4][r0];

    // ---- reload P as two 16x32 A fragments ----
    v16bf pa[2];
#pragma unroll
    for (int kc = 0; kc < 2; kc++) {
      const __bf16* prow = myp + (size_t)laneLo * PP + kc * 32;
      v8bf lo = *(const v8bf*)(prow + 16 * laneHi);
      v8bf hi = *(const v8bf*)(prow + 16 * laneHi + 8);
#pragma unroll
      for (int i = 0; i < 8; i++) { pa[kc][i] = lo[i]; pa[kc][8 + i] = hi[i]; }
    }

    // ---- acc += P x V over this wave's 128 channels (from LDS) ----
#pragma unroll
    for (int ct = 0; ct < 8; ct++) {
      int ch = h * 128 + ct * 16 + laneLo;
      const __bf16* vrow = Vb + (size_t)ch * VP;
      v16bf vb0 = ld32(vrow + 16 * laneHi);
      v16bf vb1 = ld32(vrow + 32 + 16 * laneHi);
      acc[ct] = wmma_bf16(pa[0], vb0, acc[ct]);
      acc[ct] = wmma_bf16(pa[1], vb1, acc[ct]);
    }
    __syncthreads();           // all waves done reading buf before it's restaged
  }

  // ---- epilogue: out = gamma * (acc / l) + y_p ----
  float g = gamma_p[0];
#pragma unroll
  for (int ct = 0; ct < 8; ct++) {
    int ch = h * 128 + ct * 16 + laneLo;
    size_t base = ((size_t)b * CC + ch) * NN + q0 + r * 16 + 8 * laneHi;
#pragma unroll
    for (int r0 = 0; r0 < 8; r0++) {
      out[base + r0] = g * (acc[ct][r0] / lrow[r0]) + ypf[base + r0];
    }
  }
}

// ---------------------------------------------------------------------------
// Launch
// ---------------------------------------------------------------------------
extern "C" void kernel_launch(void* const* d_in, const int* in_sizes, int n_in,
                              void* d_out, int out_size, void* d_ws, size_t ws_size,
                              hipStream_t stream) {
  const float* x     = (const float*)d_in[0];
  const float* sa_w  = (const float*)d_in[1];
  const float* ca_w1 = (const float*)d_in[2];
  const float* ca_w2 = (const float*)d_in[3];
  const float* pa_w1 = (const float*)d_in[4];
  const float* pa_w2 = (const float*)d_in[5];
  const float* q_w   = (const float*)d_in[6];
  const float* q_b   = (const float*)d_in[7];
  const float* k_w   = (const float*)d_in[8];
  const float* k_b   = (const float*)d_in[9];
  const float* v_w   = (const float*)d_in[10];
  const float* v_b   = (const float*)d_in[11];
  const float* gamma = (const float*)d_in[12];
  float* out = (float*)d_out;

  size_t off = 0;
  auto carve = [&](size_t bytes) { size_t o = off; off += (bytes + 255) & ~(size_t)255; return o; };
  char* w = (char*)d_ws;
  float*  ys   = (float*)(w + carve(sizeof(float) * BB * NN));
  float*  cavg = (float*)(w + carve(sizeof(float) * BB * CC));
  float*  cmax = (float*)(w + carve(sizeof(float) * BB * CC));
  float*  yc   = (float*)(w + carve(sizeof(float) * BB * CC));
  __bf16* zT   = (__bf16*)(w + carve(sizeof(__bf16) * (size_t)BB * NN * 512));
  __bf16* ypT  = (__bf16*)(w + carve(sizeof(__bf16) * (size_t)BB * NN * 256));
  float*  ypf  = (float*)(w + carve(sizeof(float) * (size_t)BB * CC * NN));
  __bf16* qT   = (__bf16*)(w + carve(sizeof(__bf16) * (size_t)BB * NN * 32));
  __bf16* kT   = (__bf16*)(w + carve(sizeof(__bf16) * (size_t)BB * NN * 32));
  __bf16* vm   = (__bf16*)(w + carve(sizeof(__bf16) * (size_t)BB * CC * NN));
  __bf16* wpa  = (__bf16*)(w + carve(sizeof(__bf16) * 256 * 512));
  __bf16* wq   = (__bf16*)(w + carve(sizeof(__bf16) * 32 * 256));
  __bf16* wk   = (__bf16*)(w + carve(sizeof(__bf16) * 32 * 256));
  __bf16* wv   = (__bf16*)(w + carve(sizeof(__bf16) * 256 * 256));
  (void)ws_size; (void)in_sizes; (void)n_in; (void)out_size;

  k_spatial <<<(BB * NN + 255) / 256, 256, 0, stream>>>(x, sa_w, ys);
  k_chanpool<<<BB * CC, 256, 0, stream>>>(x, cavg, cmax);
  k_camlp   <<<BB, 256, 0, stream>>>(cavg, cmax, ca_w1, ca_w2, yc);

  k_fusez<<<dim3(BB, CC / 32, NN / 32), dim3(32, 8), 0, stream>>>(x, ys, yc, zT);

  k_cvt<<<(256 * 512 + 255) / 256, 256, 0, stream>>>(pa_w1, wpa, 256 * 512,
                                                     q_w, wq, 32 * 256,
                                                     k_w, wk, 32 * 256,
                                                     v_w, wv, 256 * 256);

  k_gemm_pa<<<(BB * 256 * 16) / 8, 256, 0, stream>>>(zT, wpa, pa_w2, ypf, ypT);

  k_gemm_qkv<<<(BB * 256 * 20) / 8, 256, 0, stream>>>(ypT, wq, wk, wv, q_b, k_b, v_b, qT, kT, vm);

  k_flash<<<BB * (NN / 64), 256, 0, stream>>>(qT, kT, vm, ypf, gamma, out);
}
